// RNNFakeDetectionModel_87333864997080
// MI455X (gfx1250) — matrix operand order repaired and checked by reading, not compiled, not run on hardware
//
#include <hip/hip_runtime.h>

// ---------------- types for WMMA fragments ----------------
typedef _Float16 v16h  __attribute__((ext_vector_type(16)));
typedef _Float16 v8h   __attribute__((ext_vector_type(8)));
typedef __bf16   v16bf __attribute__((ext_vector_type(16)));
typedef __bf16   v8bf  __attribute__((ext_vector_type(8)));
typedef float    v8f   __attribute__((ext_vector_type(8)));

union AF16 { v16h  v; v8h  h[2]; };
union ABF  { v16bf v; v8bf h[2]; };

// ---------------- problem constants ----------------
#define SRL   160000   // samples per waveform
#define NFFT  400
#define HOPW  160
#define NFRM  1001     // frames per waveform
#define TPAD  1008     // frames padded to 63 tiles of 16
#define NBIN  201      // rfft bins
#define NPAD  208      // bins padded: 13 n-tiles of 16
#define KPADF 416      // fft K padded: 13 k-steps of 32
#define NMEL  40
#define MELP  48       // mel padded: 3 n-tiles
#define PKP   224      // power K padded: 7 k-steps of 32

__device__ inline __bf16 f2bf(float x) {
  unsigned u = __float_as_uint(x);
  unsigned r = (u + 0x7FFFu + ((u >> 16) & 1u)) >> 16;
  unsigned short s = (unsigned short)r;
  return __builtin_bit_cast(__bf16, s);
}

// ---------------- workspace layout ----------------
static constexpr size_t AL(size_t x) { return (x + 255) & ~(size_t)255; }
static constexpr size_t OFF_CB   = 0;
static constexpr size_t SZ_CB    = (size_t)NPAD * KPADF * 2;       // f16 cos basis^T [208][416]
static constexpr size_t OFF_SB   = AL(OFF_CB + SZ_CB);             // f16 sin basis^T
static constexpr size_t OFF_FB   = AL(OFF_SB + SZ_CB);             // bf16 melFB^T [48][224]
static constexpr size_t SZ_FB    = (size_t)MELP * PKP * 2;
static constexpr size_t OFF_DCT  = AL(OFF_FB + SZ_FB);             // f32 dct [40][16]
static constexpr size_t SZ_DCT   = 640 * 4;
static constexpr size_t OFF_HANN = AL(OFF_DCT + SZ_DCT);           // f32 hann [400]
static constexpr size_t SZ_HANN  = 400 * 4;
static constexpr size_t OFF_WIH  = AL(OFF_HANN + SZ_HANN);         // f16 W_ih^T-as-[n][k] [64][32]
static constexpr size_t SZ_WIH   = 64 * 32 * 2;
static constexpr size_t OFF_WHH  = AL(OFF_WIH + SZ_WIH);           // f16 W_hh [64][64]
static constexpr size_t SZ_WHH   = 64 * 64 * 2;
static constexpr size_t OFF_BS   = AL(OFF_WHH + SZ_WHH);           // f32 b_ih+b_hh [64]
static constexpr size_t SZ_BS    = 64 * 4;
static constexpr size_t OFF_MAX  = AL(OFF_BS + SZ_BS);             // u32 per-batch max(mel) bits [64]
static constexpr size_t SZ_MAX   = 64 * 4;
static constexpr size_t OFF_MEL  = AL(OFF_MAX + SZ_MAX);           // f32 mel [64][1008][40]
static constexpr size_t SZ_MEL   = (size_t)64 * TPAD * NMEL * 4;
static constexpr size_t OFF_MFCC = AL(OFF_MEL + SZ_MEL);           // f32 mfccT [1001][64][16]
static constexpr size_t SZ_MFCC  = (size_t)NFRM * 64 * 16 * 4;

// ======================================================================
// Kernel 0: build all constant tables on device (model constants)
// ======================================================================
__global__ __launch_bounds__(256) void setup_kernel(
    const float* __restrict__ W_ih, const float* __restrict__ W_hh,
    const float* __restrict__ b_ih, const float* __restrict__ b_hh,
    _Float16* __restrict__ cbasisT, _Float16* __restrict__ sbasisT,
    __bf16* __restrict__ fbT, float* __restrict__ dctM, float* __restrict__ hann,
    _Float16* __restrict__ wihT, _Float16* __restrict__ whhT,
    float* __restrict__ bsum, unsigned* __restrict__ maxbuf)
{
  const int i = blockIdx.x * 256 + threadIdx.x;
  const float PI = 3.14159265358979323846f;
  const float W  = 2.f * PI / 400.f;
  if (i < NPAD * KPADF) {                       // Hann-folded DFT basis, transposed [bin][k]
    int n = i / KPADF, k = i - n * KPADF;
    float c = 0.f, s = 0.f;
    if (n < NBIN && k < NFFT) {
      int r = (n * k) % 400;                    // exact phase reduction
      float arg = W * (float)r;
      float hv  = 0.5f * (1.f - cosf(W * (float)k));
      c = hv * cosf(arg);
      s = hv * sinf(arg);
    }
    cbasisT[i] = (_Float16)c;
    sbasisT[i] = (_Float16)s;
  }
  if (i < 400) hann[i] = 0.5f * (1.f - cosf(W * (float)i));
  if (i < MELP * PKP) {                         // HTK mel filterbank, transposed [mel][bin]
    int n = i / PKP, k = i - n * PKP;
    float v = 0.f;
    if (n < NMEL && k < NBIN) {
      float freq = 40.f * (float)k;             // 8000/200 spacing
      float mmax = 2595.f * log10f(1.f + 8000.f / 700.f);
      float m0 = mmax * (float)(n)     / 41.f;
      float m1 = mmax * (float)(n + 1) / 41.f;
      float m2 = mmax * (float)(n + 2) / 41.f;
      float f0 = 700.f * (powf(10.f, m0 / 2595.f) - 1.f);
      float f1 = 700.f * (powf(10.f, m1 / 2595.f) - 1.f);
      float f2 = 700.f * (powf(10.f, m2 / 2595.f) - 1.f);
      float down = (freq - f0) / (f1 - f0);
      float up   = (f2 - freq) / (f2 - f1);
      v = fmaxf(0.f, fminf(down, up));
    }
    fbT[i] = f2bf(v);
  }
  if (i < 640) {                                // DCT-II ortho [mel][coef] stride 16
    int mI = i >> 4, c = i & 15;
    float v = 0.f;
    if (c < 13) {
      v = 2.f * cosf(PI / 40.f * ((float)mI + 0.5f) * (float)c) * 0.1118033988749895f;
      if (c == 0) v *= 0.7071067811865476f;
    }
    dctM[i] = v;
  }
  if (i < 64 * 32) {                            // wihT[n][k] = W_ih[n][k], K padded to 32
    int n = i >> 5, k = i & 31;
    wihT[i] = (_Float16)(k < 13 ? W_ih[n * 13 + k] : 0.f);
  }
  if (i < 64 * 64) whhT[i] = (_Float16)W_hh[i]; // whhT[n][k] = W_hh[n][k]
  if (i < 64) { bsum[i] = b_ih[i] + b_hh[i]; maxbuf[i] = 0u; }
}

// ======================================================================
// Kernel 1: framing + windowed DFT (f16 WMMA) + power + mel (bf16 WMMA)
// one block = one batch * one 16-frame tile
// ======================================================================
__global__ __launch_bounds__(256) void spectro_mel_kernel(
    const float* __restrict__ wav, const _Float16* __restrict__ cbasisT,
    const _Float16* __restrict__ sbasisT, const __bf16* __restrict__ fbT,
    const float* __restrict__ hann, float* __restrict__ mel,
    unsigned* __restrict__ maxbuf)
{
  __shared__ __align__(32) _Float16 As[16][KPADF];   // windowed frames, f16
  __shared__ __align__(32) __bf16   Ps[16][PKP];     // power spectrum, bf16
  __shared__ unsigned lmax;
  const int bx  = blockIdx.x;
  const int b   = bx / 63;
  const int t0  = (bx - b * 63) * 16;
  const int tid = threadIdx.x;
  const int lane = tid & 31, wave = tid >> 5;
  const int g = lane >> 4, ln = lane & 15;
  if (tid == 0) lmax = 0u;
  for (int i = tid; i < 16 * PKP; i += 256) ((__bf16*)Ps)[i] = f2bf(0.f);

  // build A tile: reflect-padded, Hann-windowed frames (uniform trip count: no divergence)
  const float* xb = wav + (size_t)b * SRL;
  for (int i = tid; i < 16 * KPADF; i += 256) {
    int row = i / KPADF, n = i - row * KPADF;
    float v = 0.f;
    if (n < NFFT) {
      int idx = (t0 + row) * HOPW + n - 200;
      if (idx < 0) idx = -idx;
      if (idx >= SRL) idx = 2 * SRL - 2 - idx;
      v = xb[idx] * hann[n];
    }
    As[row][n] = (_Float16)v;
  }
  __syncthreads();

  // DFT: C[16,16] tiles over 13 n-tiles x 13 k-steps, cos & sin simultaneously
  for (int job = wave; job < 13; job += 8) {
    const _Float16* cb = cbasisT + (size_t)(job * 16 + ln) * KPADF;
    const _Float16* sb = sbasisT + (size_t)(job * 16 + ln) * KPADF;
    v8f accC = {}, accS = {};
    for (int ks = 0; ks < 13; ++ks) {
      const int k0 = ks * 32;
      AF16 a;
      a.h[0] = *(const v8h*)&As[ln][k0 + 16 * g];
      a.h[1] = *(const v8h*)&As[ln][k0 + 16 * g + 8];
      v16h bc = *(const v16h*)(cb + k0 + 16 * g);
      v16h bs = *(const v16h*)(sb + k0 + 16 * g);
      accC = __builtin_amdgcn_wmma_f32_16x16x32_f16(false, a.v, false, bc, (short)0, accC, false, false);
      accS = __builtin_amdgcn_wmma_f32_16x16x32_f16(false, a.v, false, bs, (short)0, accS, false, false);
    }
    const int n = job * 16 + ln;
#pragma unroll
    for (int r = 0; r < 8; ++r) {
      float pw = accC[r] * accC[r] + accS[r] * accS[r];   // |X|^2 = re^2 + im^2
      Ps[r + 8 * g][n] = f2bf(pw);
    }
  }
  __syncthreads();

  // mel: [16,224] x [224,48] in bf16 WMMA; waves 0..2
  float wmax = 0.f;
  if (wave < 3) {
    const __bf16* fb = fbT + (size_t)(wave * 16 + ln) * PKP;
    v8f acc = {};
    for (int ks = 0; ks < 7; ++ks) {
      const int k0 = ks * 32;
      ABF a;
      a.h[0] = *(const v8bf*)&Ps[ln][k0 + 16 * g];
      a.h[1] = *(const v8bf*)&Ps[ln][k0 + 16 * g + 8];
      v16bf bb = *(const v16bf*)(fb + k0 + 16 * g);
      acc = __builtin_amdgcn_wmma_f32_16x16x32_bf16(false, a.v, false, bb, (short)0, acc, false, false);
    }
    const int n = wave * 16 + ln;
#pragma unroll
    for (int r = 0; r < 8; ++r) {
      int t = t0 + r + 8 * g;
      if (t < NFRM && n < NMEL) {
        float v = acc[r];
        mel[((size_t)b * TPAD + t) * NMEL + n] = v;
        wmax = fmaxf(wmax, v);
      }
    }
  }
  // mel >= 0 so float bits compare as uint: exact, order-independent max
  atomicMax(&lmax, __float_as_uint(wmax));
  __syncthreads();
  if (tid == 0) atomicMax(maxbuf + b, lmax);
}

// ======================================================================
// Kernel 2: dB (with per-sample top_db clamp) + DCT -> mfccT[t][b][16]
// ======================================================================
__global__ __launch_bounds__(256) void db_dct_kernel(
    const float* __restrict__ mel, const unsigned* __restrict__ maxbuf,
    const float* __restrict__ dctM, float* __restrict__ mfccT)
{
  int i = blockIdx.x * 256 + threadIdx.x;
  if (i >= 64 * NFRM) return;
  int b = i / NFRM, t = i - b * NFRM;
  float mm  = __uint_as_float(maxbuf[b]);
  float thr = 10.f * log10f(fmaxf(mm, 1e-10f)) - 80.f;
  const float* mr = mel + ((size_t)b * TPAD + t) * NMEL;
  float db[NMEL];
#pragma unroll 8
  for (int m = 0; m < NMEL; ++m)
    db[m] = fmaxf(10.f * log10f(fmaxf(mr[m], 1e-10f)), thr);
  float* o = mfccT + ((size_t)t * 64 + b) * 16;
  for (int c = 0; c < 13; ++c) {
    float s = 0.f;
    for (int m = 0; m < NMEL; ++m) s += db[m] * dctM[m * 16 + c];
    o[c] = s;
  }
}

// ======================================================================
// Kernel 3: persistent single-WG tanh RNN (f16 WMMA, weights resident in
// VGPRs, LDS ping-pong hidden state) + MLP head + softmax
// ======================================================================
__global__ __launch_bounds__(256) void rnn_head_kernel(
    const float* __restrict__ mfccT, const _Float16* __restrict__ wihT,
    const _Float16* __restrict__ whhT, const float* __restrict__ bsum,
    const float* __restrict__ W1, const float* __restrict__ b1,
    const float* __restrict__ W2, const float* __restrict__ b2,
    float* __restrict__ out)
{
  __shared__ __align__(32) _Float16 hbuf[2][64][64];
  __shared__ __align__(32) _Float16 Xs[64][32];
  __shared__ float x1[64][32];
  const int tid = threadIdx.x;
  const int lane = tid & 31, wave = tid >> 5;
  const int g = lane >> 4, ln = lane & 15;
  const int mi = wave & 3;              // batch tile
  const int ni0 = (wave >> 2) * 2;      // two hidden tiles per wave

  for (int i = tid; i < 64 * 64; i += 256) ((_Float16*)hbuf[0])[i] = (_Float16)0.f;
  for (int i = tid; i < 64 * 32; i += 256) ((_Float16*)Xs)[i] = (_Float16)0.f;

  // preload constant B fragments (weights) into VGPRs once
  v16h bih[2], bhh0[2], bhh1[2];
  float bias[2];
#pragma unroll
  for (int q = 0; q < 2; ++q) {
    int n = (ni0 + q) * 16 + ln;
    bih[q]  = *(const v16h*)(wihT + n * 32 + 16 * g);
    bhh0[q] = *(const v16h*)(whhT + n * 64 + 16 * g);
    bhh1[q] = *(const v16h*)(whhT + n * 64 + 32 + 16 * g);
    bias[q] = bsum[n];
  }
  __syncthreads();

  int p = 0;
  const int m = mi * 16 + ln;
  for (int t = 0; t < NFRM; ++t) {
    for (int i = tid; i < 64 * 13; i += 256) {       // stage x_t as f16 [64][32]
      int bb = i / 13, j = i - bb * 13;
      Xs[bb][j] = (_Float16)mfccT[((size_t)t * 64 + bb) * 16 + j];
    }
    __syncthreads();
    AF16 ax, ah0, ah1;
    ax.h[0]  = *(const v8h*)&Xs[m][16 * g];
    ax.h[1]  = *(const v8h*)&Xs[m][16 * g + 8];
    ah0.h[0] = *(const v8h*)&hbuf[p][m][16 * g];
    ah0.h[1] = *(const v8h*)&hbuf[p][m][16 * g + 8];
    ah1.h[0] = *(const v8h*)&hbuf[p][m][32 + 16 * g];
    ah1.h[1] = *(const v8h*)&hbuf[p][m][32 + 16 * g + 8];
#pragma unroll
    for (int q = 0; q < 2; ++q) {
      v8f acc = {};
      acc = __builtin_amdgcn_wmma_f32_16x16x32_f16(false, ax.v,  false, bih[q],  (short)0, acc, false, false);
      acc = __builtin_amdgcn_wmma_f32_16x16x32_f16(false, ah0.v, false, bhh0[q], (short)0, acc, false, false);
      acc = __builtin_amdgcn_wmma_f32_16x16x32_f16(false, ah1.v, false, bhh1[q], (short)0, acc, false, false);
      int n = (ni0 + q) * 16 + ln;
#pragma unroll
      for (int r = 0; r < 8; ++r) {
        float hv = tanhf(acc[r] + bias[q]);
        hbuf[1 - p][mi * 16 + r + 8 * g][n] = (_Float16)hv;
      }
    }
    p ^= 1;
    __syncthreads();
  }

  // head: relu(h @ W1^T + b1) @ W2^T + b2 -> softmax
  for (int i = tid; i < 64 * 32; i += 256) {
    int bb = i >> 5, d = i & 31;
    float s = b1[d];
    for (int j = 0; j < 64; ++j) s += (float)hbuf[p][bb][j] * W1[d * 64 + j];
    x1[bb][d] = fmaxf(s, 0.f);
  }
  __syncthreads();
  if (tid < 64) {
    float l0 = b2[0], l1 = b2[1];
    for (int d = 0; d < 32; ++d) { l0 += x1[tid][d] * W2[d]; l1 += x1[tid][d] * W2[32 + d]; }
    float mx = fmaxf(l0, l1);
    float e0 = __expf(l0 - mx), e1 = __expf(l1 - mx);
    float inv = 1.f / (e0 + e1);
    out[tid * 2 + 0] = e0 * inv;
    out[tid * 2 + 1] = e1 * inv;
  }
}

// ======================================================================
extern "C" void kernel_launch(void* const* d_in, const int* in_sizes, int n_in,
                              void* d_out, int out_size, void* d_ws, size_t ws_size,
                              hipStream_t stream) {
  (void)in_sizes; (void)n_in; (void)out_size; (void)ws_size;
  const float* wav  = (const float*)d_in[0];
  const float* W_ih = (const float*)d_in[1];
  const float* W_hh = (const float*)d_in[2];
  const float* b_ih = (const float*)d_in[3];
  const float* b_hh = (const float*)d_in[4];
  const float* W1   = (const float*)d_in[5];
  const float* b1   = (const float*)d_in[6];
  const float* W2   = (const float*)d_in[7];
  const float* b2   = (const float*)d_in[8];
  char* ws = (char*)d_ws;

  _Float16* cbasisT = (_Float16*)(ws + OFF_CB);
  _Float16* sbasisT = (_Float16*)(ws + OFF_SB);
  __bf16*   fbT     = (__bf16*)  (ws + OFF_FB);
  float*    dctM    = (float*)   (ws + OFF_DCT);
  float*    hann    = (float*)   (ws + OFF_HANN);
  _Float16* wihT    = (_Float16*)(ws + OFF_WIH);
  _Float16* whhT    = (_Float16*)(ws + OFF_WHH);
  float*    bsum    = (float*)   (ws + OFF_BS);
  unsigned* maxbuf  = (unsigned*)(ws + OFF_MAX);
  float*    mel     = (float*)   (ws + OFF_MEL);
  float*    mfccT   = (float*)   (ws + OFF_MFCC);

  setup_kernel<<<338, 256, 0, stream>>>(W_ih, W_hh, b_ih, b_hh, cbasisT, sbasisT,
                                        fbT, dctM, hann, wihT, whhT, bsum, maxbuf);
  spectro_mel_kernel<<<64 * 63, 256, 0, stream>>>(wav, cbasisT, sbasisT, fbT, hann, mel, maxbuf);
  db_dct_kernel<<<(64 * NFRM + 255) / 256, 256, 0, stream>>>(mel, maxbuf, dctM, mfccT);
  rnn_head_kernel<<<1, 256, 0, stream>>>(mfccT, wihT, whhT, bsum, W1, b1, W2, b2, (float*)d_out);
}
